// AttentionRNN_22677427323565
// MI455X (gfx1250) — compile-verified, matrix operand order checked by reading, not run on hardware
//
#include <hip/hip_runtime.h>
#include <math.h>

// ---------------------------------------------------------------------------
// Graves attention-RNN, persistent-kernel, batch-partitioned (no cross-WG sync).
// 16 workgroups x 16 batch rows; activations in LDS (bf16), weights prepacked
// to bf16 in workspace (L2-resident); all GEMMs via v_wmma_f32_16x16x32_bf16.
// k-loops kept rolled (#pragma unroll 1) to avoid VGPR spills of B-fragments.
// Branch-free tanh/sigmoid with v_rcp_f32 keep the serial path short.
// ---------------------------------------------------------------------------

typedef __attribute__((ext_vector_type(16))) __bf16        v16bf;
typedef __attribute__((ext_vector_type(8)))  float         v8f;
typedef __attribute__((ext_vector_type(4)))  unsigned int  u32x4;

#define BATCH 256
#define SEQ   1024
#define EMB   128
#define HID   256
#define MIX   10
#define WLEN  32
#define OUTW  121
#define K1    416      // [wv(128) | x(3) | h1(256)] padded 387->416
#define K2    640      // [h1(256) | wv(128) | h2(256)]
#define K3    640      // [h2(256) | wv(128) | h3(256)]
#define KO    256      // h3
#define G4H   1024     // 4*H gates

// workspace offsets (ushort elements)
#define OFF_WC1 0
#define OFF_WA  (OFF_WC1 + G4H * K1)          // 32 x 416 (30 rows used)
#define OFF_WC2 (OFF_WA  + 32  * K1)
#define OFF_WC3 (OFF_WC2 + G4H * K2)
#define OFF_WO  (OFF_WC3 + G4H * K3)          // 128 x 256 (121 rows used)
#define WS_TOT  (OFF_WO  + 128 * KO)

__device__ __forceinline__ unsigned short f2bf(float f) {
    unsigned int u = __float_as_uint(f);
    unsigned int r = u + 0x7FFFu + ((u >> 16) & 1u);   // round-to-nearest-even
    return (unsigned short)(r >> 16);
}
// hardware reciprocal (v_rcp_f32, ~1 ULP) — avoids IEEE div_scale/NR sequence
__device__ __forceinline__ float rcpf(float x) { return __builtin_amdgcn_rcpf(x); }
__device__ __forceinline__ float sigmf(float x) { return rcpf(1.f + __expf(-x)); }
// branch-free tanh: exact at +/-inf because __expf saturates (inf -> 1, 0 -> -1)
__device__ __forceinline__ float tanhfast(float x) {
    return 1.f - 2.f * rcpf(1.f + __expf(2.f * x));
}
__device__ __forceinline__ float softplusf(float x) { return (x > 20.f) ? x : log1pf(__expf(x)); }

// ---------------------------------------------------------------------------
// Prepack combined bf16 weight matrices:  Wc = [W_ih | W_hh] (row-major, K padded)
// ---------------------------------------------------------------------------
__global__ void prep_weights(const float* __restrict__ Wih1, const float* __restrict__ Whh1,
                             const float* __restrict__ Wih2, const float* __restrict__ Whh2,
                             const float* __restrict__ Wih3, const float* __restrict__ Whh3,
                             const float* __restrict__ Watt, const float* __restrict__ Wout,
                             unsigned short* __restrict__ ws) {
    for (int i = blockIdx.x * blockDim.x + threadIdx.x; i < WS_TOT;
         i += gridDim.x * blockDim.x) {
        float v = 0.f;
        if (i < OFF_WA) {                       // Wc1: 1024 x 416, [W_ih1(131)|W_hh1(256)|pad]
            int g = i / K1, k = i % K1;
            if (k < 131)       v = Wih1[g * 131 + k];
            else if (k < 387)  v = Whh1[g * 256 + (k - 131)];
        } else if (i < OFF_WC2) {               // Wa: 32 x 416, rows>=30 & k>=387 zero
            int j = i - OFF_WA; int n = j / K1, k = j % K1;
            if (n < 30 && k < 387) v = Watt[n * 387 + k];
        } else if (i < OFF_WC3) {               // Wc2: 1024 x 640, [W_ih2(384)|W_hh2(256)]
            int j = i - OFF_WC2; int g = j / K2, k = j % K2;
            v = (k < 384) ? Wih2[g * 384 + k] : Whh2[g * 256 + (k - 384)];
        } else if (i < OFF_WO) {                // Wc3
            int j = i - OFF_WC3; int g = j / K3, k = j % K3;
            v = (k < 384) ? Wih3[g * 384 + k] : Whh3[g * 256 + (k - 384)];
        } else {                                // Wo: 128 x 256, rows>=121 zero
            int j = i - OFF_WO; int n = j / KO, k = j % KO;
            if (n < OUTW) v = Wout[n * KO + k];
        }
        ws[i] = f2bf(v);
    }
}

// ---------------------------------------------------------------------------
// WMMA tile helpers.  A: 16 rows (batch) x K from LDS. B: rows of W (gate idx)
// from global bf16.  Layouts per CDNA5 ISA 7.12.2 (lane halves split K at 8/24).
// ---------------------------------------------------------------------------
struct Frag { union { u32x4 u[2]; v16bf v; }; };

template <int NG>
__device__ __forceinline__ void gemm_rows(const unsigned short* __restrict__ Abase, int astride,
                                          const unsigned short* __restrict__ W, int Kpad, int ksteps,
                                          const int* nbase, const float* bias,
                                          int lane, v8f* acc) {
#pragma unroll
    for (int g = 0; g < NG; ++g) {
#pragma unroll
        for (int e = 0; e < 8; ++e) acc[g][e] = bias[g];
    }
    const int m  = lane & 15;
    const int kh = (lane >> 4) << 3;   // lanes 16-31 take K+8..15 / K+24..31
    const unsigned short* pa = Abase + m * astride + kh;
    const unsigned short* pw = W + m * (size_t)Kpad + kh;
#pragma unroll 1                        // keep rolled: bounds live B-frags, no spills
    for (int kt = 0; kt < ksteps; ++kt) {
        const int kb = kt * 32;
        // load all B fragments first -> single clause + one wait, then WMMA burst
        Frag b[NG];
#pragma unroll
        for (int g = 0; g < NG; ++g) {
            const unsigned short* pb = pw + (size_t)nbase[g] * Kpad + kb;
            b[g].u[0] = *(const u32x4*)(pb);
            b[g].u[1] = *(const u32x4*)(pb + 16);
        }
        Frag a;
        a.u[0] = *(const u32x4*)(pa + kb);
        a.u[1] = *(const u32x4*)(pa + kb + 16);
#pragma unroll
        for (int g = 0; g < NG; ++g) {
            acc[g] = __builtin_amdgcn_wmma_f32_16x16x32_bf16(
                         false, a.v, false, b[g].v, (short)0, acc[g], false, false);
        }
    }
}

// LSTM pointwise: gates (i,f,g,o) fragments -> new c (register-resident) and
// h written as bf16 into one or two LDS A-slices.
__device__ __forceinline__ void lstm_act(const v8f* acc, v8f& c,
                                         unsigned short* d1, int s1, int c1,
                                         unsigned short* d2, int s2, int c2,
                                         int lane) {
    const int mh = (lane >> 4) << 3;
#pragma unroll
    for (int e = 0; e < 8; ++e) {
        int m = e + mh;
        float ii = sigmf(acc[0][e]);
        float ff = sigmf(acc[1][e]);
        float gg = tanhfast(acc[2][e]);
        float oo = sigmf(acc[3][e]);
        float cn = ff * c[e] + ii * gg;
        c[e] = cn;
        unsigned short hb = f2bf(oo * tanhfast(cn));
        d1[m * s1 + c1] = hb;
        if (d2) d2[m * s2 + c2] = hb;
    }
}

// ---------------------------------------------------------------------------
// Persistent RNN kernel: blockIdx.x = batch group (16 rows), 512 thr = 16 waves
// ---------------------------------------------------------------------------
__launch_bounds__(512, 1)
__global__ void rnn_kernel(const float* __restrict__ traj, const int* __restrict__ words,
                           const int* __restrict__ wlens, const float* __restrict__ emb,
                           const float* __restrict__ b1, const float* __restrict__ b2,
                           const float* __restrict__ b3, const float* __restrict__ batt,
                           const float* __restrict__ bout,
                           const unsigned short* __restrict__ ws, float* __restrict__ out) {
    __shared__ __align__(16) unsigned short A1[16 * K1];
    __shared__ __align__(16) unsigned short A2[16 * K2];
    __shared__ __align__(16) unsigned short A3[16 * K3];
    __shared__ float attg[16 * 32];
    __shared__ float alph[16 * MIX], betr[16 * MIX], kap[16 * MIX];  // betr = 1/beta
    __shared__ float phi[16 * WLEN];
    __shared__ float pinv[16];

    const int tid  = threadIdx.x;
    const int wave = tid >> 5;
    const int lane = tid & 31;
    const int ncl  = lane & 15;
    const int bg   = blockIdx.x;                  // 16 groups of 16 batch rows
    const int hc   = wave * 16 + ncl;             // hidden column owned by this lane

    // zero-init recurrent LDS state
    for (int i = tid; i < 16 * K1;  i += 512) A1[i] = 0;
    for (int i = tid; i < 16 * K2;  i += 512) A2[i] = 0;
    for (int i = tid; i < 16 * K3;  i += 512) A3[i] = 0;
    for (int i = tid; i < 16 * MIX; i += 512) kap[i] = 0.f;

    v8f c1s, c2s, c3s;
#pragma unroll
    for (int e = 0; e < 8; ++e) { c1s[e] = 0.f; c2s[e] = 0.f; c3s[e] = 0.f; }

    // hoist biases (constant across steps) into registers
    const int nb[4] = { wave * 16, 256 + wave * 16, 512 + wave * 16, 768 + wave * 16 };
    float bias1[4], bias2[4], bias3[4];
#pragma unroll
    for (int g = 0; g < 4; ++g) {
        bias1[g] = b1[g * 256 + hc];
        bias2[g] = b2[g * 256 + hc];
        bias3[g] = b3[g * 256 + hc];
    }
    const float biasA = (wave < 2 && (wave * 16 + ncl) < 30)  ? batt[wave * 16 + ncl] : 0.f;
    const float biasO = (wave < 8 && (wave * 16 + ncl) < OUTW) ? bout[wave * 16 + ncl] : 0.f;

    __syncthreads();

    for (int t = 0; t < SEQ; ++t) {
        // ---- write x_t into A1[:,128:131] -------------------------------
        if (tid < 48) {
            int r = tid / 3, j = tid - r * 3;
            A1[r * K1 + 128 + j] = f2bf(traj[((size_t)(bg * 16 + r) * SEQ + t) * 3 + j]);
        }
        __syncthreads();

        // ---- LSTM1 gates: A1 @ Wc1^T  (K=416) ---------------------------
        v8f acc[4];
        gemm_rows<4>(A1, K1, ws + OFF_WC1, K1, K1 / 32, nb, bias1, lane, acc);
        __syncthreads();
        lstm_act(acc, c1s, A1, K1, 131 + hc, A2, K2, hc, lane);   // h1 -> A1, A2
        __syncthreads();

        // ---- attention projection: A1 @ Wa^T (30->32 cols) --------------
        if (wave < 2) {
            v8f aacc[1]; int nba[1] = { wave * 16 }; float bba[1] = { biasA };
            gemm_rows<1>(A1, K1, ws + OFF_WA, K1, K1 / 32, nba, bba, lane, aacc);
            const int mh = (lane >> 4) << 3;
#pragma unroll
            for (int e = 0; e < 8; ++e)
                attg[(e + mh) * 32 + wave * 16 + ncl] = aacc[0][e];
        }
        __syncthreads();

        // ---- Gaussian mixture attention (f32 VALU, per-batch-row) -------
        {
            const int row = tid >> 4, sub = tid & 15;
            const bool act = tid < 256;
            const int  b   = bg * 16 + row;
            if (act && sub < MIX) {
                alph[row * MIX + sub] = softplusf(attg[row * 32 + sub]);
                float be = softplusf(attg[row * 32 + 10 + sub]);
                betr[row * MIX + sub] = rcpf((be < 0.01f) ? 0.01f : be);
                kap[row * MIX + sub] += softplusf(attg[row * 32 + 20 + sub]) * 0.04f;
            }
            __syncthreads();
            if (act) {
                int wl = wlens[b];
#pragma unroll
                for (int q = 0; q < 2; ++q) {
                    int l = sub + q * 16;
                    float s = 0.f;
#pragma unroll
                    for (int m = 0; m < MIX; ++m) {
                        float d = kap[row * MIX + m] - (float)l;
                        s += alph[row * MIX + m] * __expf(-d * d * betr[row * MIX + m]);
                    }
                    phi[row * 32 + l] = (l < wl) ? s : 0.f;
                }
            }
            __syncthreads();
            if (act && sub == 0) {
                float s = 0.f;
                for (int l = 0; l < 32; ++l) s += phi[row * 32 + l];
                pinv[row] = rcpf(s + 1e-8f);
            }
            __syncthreads();
            if (act) {
                float inv = pinv[row];
                for (int e = sub * 8; e < sub * 8 + 8; ++e) {
                    float s = 0.f;
                    for (int l = 0; l < 32; ++l)
                        s += phi[row * 32 + l] * emb[(size_t)words[b * 32 + l] * EMB + e];
                    unsigned short wb = f2bf(s * inv);
                    A1[row * K1 + e]       = wb;        // wv for LSTM1/att next step
                    A2[row * K2 + 256 + e] = wb;        // wv for LSTM2
                    A3[row * K3 + 256 + e] = wb;        // wv for LSTM3
                }
            }
        }
        __syncthreads();

        // ---- LSTM2: A2=[h1|wv|h2_prev] @ Wc2^T  (K=640) -----------------
        gemm_rows<4>(A2, K2, ws + OFF_WC2, K2, K2 / 32, nb, bias2, lane, acc);
        __syncthreads();
        lstm_act(acc, c2s, A2, K2, 384 + hc, A3, K3, hc, lane);   // h2 -> A2, A3
        __syncthreads();

        // ---- LSTM3: A3=[h2|wv|h3_prev] @ Wc3^T  (K=640) -----------------
        gemm_rows<4>(A3, K3, ws + OFF_WC3, K3, K3 / 32, nb, bias3, lane, acc);
        __syncthreads();
        lstm_act(acc, c3s, A3, K3, 384 + hc, (unsigned short*)0, 0, 0, lane); // h3 -> A3
        __syncthreads();

        // ---- output: h3 @ Wout^T (K=256, N padded 121->128) -------------
        if (wave < 8) {
            v8f oacc[1]; int nbo[1] = { wave * 16 }; float bbo[1] = { biasO };
            gemm_rows<1>(A3 + 384, K3, ws + OFF_WO, KO, KO / 32, nbo, bbo, lane, oacc);
            const int nc = wave * 16 + ncl;
            if (nc < OUTW) {
                const int mh = (lane >> 4) << 3;
#pragma unroll
                for (int e = 0; e < 8; ++e) {
                    int m = e + mh;
                    out[((size_t)(bg * 16 + m) * SEQ + t) * OUTW + nc] = oacc[0][e];
                }
            }
        }
        __syncthreads();
    }
}

// ---------------------------------------------------------------------------
extern "C" void kernel_launch(void* const* d_in, const int* in_sizes, int n_in,
                              void* d_out, int out_size, void* d_ws, size_t ws_size,
                              hipStream_t stream) {
    const float* traj  = (const float*)d_in[0];
    const int*   words = (const int*)  d_in[1];
    const int*   wlens = (const int*)  d_in[2];
    const float* emb   = (const float*)d_in[3];
    const float* Wih1  = (const float*)d_in[4];
    const float* Whh1  = (const float*)d_in[5];
    const float* b1    = (const float*)d_in[6];
    const float* Wih2  = (const float*)d_in[7];
    const float* Whh2  = (const float*)d_in[8];
    const float* b2    = (const float*)d_in[9];
    const float* Wih3  = (const float*)d_in[10];
    const float* Whh3  = (const float*)d_in[11];
    const float* b3    = (const float*)d_in[12];
    const float* Watt  = (const float*)d_in[13];
    const float* batt  = (const float*)d_in[14];
    const float* Wout  = (const float*)d_in[15];
    const float* bout  = (const float*)d_in[16];

    unsigned short* ws = (unsigned short*)d_ws;
    float* outp = (float*)d_out;

    prep_weights<<<1024, 256, 0, stream>>>(Wih1, Whh1, Wih2, Whh2, Wih3, Whh3,
                                           Watt, Wout, ws);
    rnn_kernel<<<16, 512, 0, stream>>>(traj, words, wlens, emb,
                                       b1, b2, b3, batt, bout, ws, outp);
}
